// ResidualBlock_4131758538785
// MI455X (gfx1250) — matrix operand order repaired and checked by reading, not compiled, not run on hardware
//
#include <hip/hip_runtime.h>
#include <hip/hip_bf16.h>
#include <math.h>

// Problem dims (compile-time constants from the reference)
#define D_MODEL 1024
#define D_INNER 2048
#define D_CONV  4
#define DT_RANK 64
#define D_STATE 16
#define SEQ_L   2048
#define XPROJ_N (DT_RANK + 2 * D_STATE) // 96
#define XPROJ_PAD 128                   // padded to a full 4-wave x NT=2 block

typedef __attribute__((ext_vector_type(16))) __bf16        v16bf;
typedef __attribute__((ext_vector_type(8)))  float         v8f;
typedef __attribute__((ext_vector_type(4)))  unsigned int  u32x4;
typedef __attribute__((ext_vector_type(2)))  unsigned int  u32x2;

union Frag16 { v16bf v; u32x4 q[2]; };

__device__ __forceinline__ unsigned short f2bf(float f) {
  // round-to-nearest-even fp32 -> bf16
  unsigned u = __builtin_bit_cast(unsigned, f);
  unsigned r = u + 0x7FFFu + ((u >> 16) & 1u);
  return (unsigned short)(r >> 16);
}
__device__ __forceinline__ float silu_f(float x)     { return x * (1.0f / (1.0f + __expf(-x))); }
__device__ __forceinline__ float softplus_f(float x) { return (x > 20.0f) ? x : log1pf(__expf(x)); }

// ---------------------------------------------------------------------------
// fp32 -> bf16 elementwise convert with optional zero tail padding
// ---------------------------------------------------------------------------
__global__ __launch_bounds__(256) void f32_to_bf16_pad_kernel(
    const float* __restrict__ in, unsigned short* __restrict__ out,
    int n_src, int n_total) {
  int i = blockIdx.x * 256 + threadIdx.x;
  if (i < n_total) out[i] = (i < n_src) ? f2bf(in[i]) : (unsigned short)0;
}

// ---------------------------------------------------------------------------
// RMSNorm over D_MODEL=1024, one block (256 thr) per row, emit bf16
// ---------------------------------------------------------------------------
__global__ __launch_bounds__(256) void rmsnorm_bf16_kernel(
    const float* __restrict__ x, const float* __restrict__ w,
    unsigned short* __restrict__ out) {
  __shared__ float red[8];
  const int row = blockIdx.x;
  const int t   = threadIdx.x;
  const float4 v = reinterpret_cast<const float4*>(x + (size_t)row * D_MODEL)[t];
  float s = v.x * v.x + v.y * v.y + v.z * v.z + v.w * v.w;
#pragma unroll
  for (int m = 16; m >= 1; m >>= 1) s += __shfl_xor(s, m, 32);
  if ((t & 31) == 0) red[t >> 5] = s;
  __syncthreads();
  float tot = 0.0f;
#pragma unroll
  for (int i = 0; i < 8; ++i) tot += red[i];
  const float rstd = rsqrtf(tot * (1.0f / D_MODEL) + 1e-5f);
  const float4 g = reinterpret_cast<const float4*>(w)[t];
  u32x2 p;
  p.x = (unsigned)f2bf(v.x * rstd * g.x) | ((unsigned)f2bf(v.y * rstd * g.y) << 16);
  p.y = (unsigned)f2bf(v.z * rstd * g.z) | ((unsigned)f2bf(v.w * rstd * g.w) << 16);
  *reinterpret_cast<u32x2*>(out + (size_t)row * D_MODEL + t * 4) = p;
}

// ---------------------------------------------------------------------------
// Generic GEMM: C[M,N] = A[M,K] * B[N,K]^T via v_wmma_f32_16x16x32_bf16.
// A: bf16 row-major (lda), B: bf16 row-major N x K (ldb) i.e. weights as stored.
// Block = 4 waves; each wave owns a 16 x (16*NT) tile.
// K loop is software-pipelined: unrolled x2 (K % 64 == 0 for all our shapes)
// with ping-pong fragment register sets so loads for step k+32 are in flight
// while the WMMAs for step k issue (avoids full s_wait_loadcnt 0x0 barriers).
// ALL tiles are valid by construction (shapes padded) -> no divergent branches
// around WMMA, EXEC stays all-ones as the ISA requires.
// EPI: 0 = plain, 1 = softplus(c + bias[n]), 2 = c + resid.
// Fragment layouts per CDNA5 ISA 7.12.2 (wave32):
//   A 16x32 bf16 : lane (half=l>>4, m=l&15): K chunks [k0+8h, +8) and [k0+16+8h, +8)
//   B 32x16 bf16 : lane (half, n=l&15): K chunks [k0+8h, +8) and [k0+16+8h, +8)
//   C 16x16 f32  : elem j -> row j+8h, col l&15
// ---------------------------------------------------------------------------
template <int NT, int EPI, bool WRITE_BF>
__global__ __launch_bounds__(128) void gemm_wmma_bf16(
    const unsigned short* __restrict__ A, int lda,
    const unsigned short* __restrict__ B, int ldb,
    float* __restrict__ C, int ldc,
    unsigned short* __restrict__ Cbf,
    int K,
    const float* __restrict__ bias,
    const float* __restrict__ resid, int ldr) {
  const int lane = threadIdx.x;
  // force wave index scalar (uniform): blockDim.x == 32 by construction
  const int wave = __builtin_amdgcn_readfirstlane(threadIdx.y);
  const int half = lane >> 4;
  const int l16  = lane & 15;
  const int m0   = blockIdx.x << 4;
  const int n0   = (blockIdx.y * 4 + wave) * (16 * NT);

  v8f acc[NT];
#pragma unroll
  for (int t = 0; t < NT; ++t)
#pragma unroll
    for (int j = 0; j < 8; ++j) acc[t][j] = 0.0f;

  const unsigned short* Arow = A + (size_t)(m0 + l16) * lda;
  const unsigned short* Brow[NT];
#pragma unroll
  for (int t = 0; t < NT; ++t)
    Brow[t] = B + (size_t)(n0 + t * 16 + l16) * ldb;

  Frag16 a0, a1, b0[NT], b1[NT];

  auto load_set = [&](int k, Frag16& a, Frag16 (&b)[NT]) {
    a.q[0] = *reinterpret_cast<const u32x4*>(Arow + k + 8 * half);
    a.q[1] = *reinterpret_cast<const u32x4*>(Arow + k + 16 + 8 * half);
#pragma unroll
    for (int t = 0; t < NT; ++t) {
      const unsigned short* bp = Brow[t] + k + 8 * half;
      b[t].q[0] = *reinterpret_cast<const u32x4*>(bp);
      b[t].q[1] = *reinterpret_cast<const u32x4*>(bp + 16);
    }
  };

  // prologue: preload k = 0
  load_set(0, a0, b0);

  int k0 = 0;
  for (; k0 < K - 64; k0 += 64) {
    load_set(k0 + 32, a1, b1); // in flight while a0/b0 WMMAs issue
    __builtin_prefetch(Arow + k0 + 128, 0, 3);
#pragma unroll
    for (int t = 0; t < NT; ++t)
      acc[t] = __builtin_amdgcn_wmma_f32_16x16x32_bf16(
          false, a0.v, false, b0[t].v, (short)0, acc[t], false, false);
    load_set(k0 + 64, a0, b0); // in flight while a1/b1 WMMAs issue
#pragma unroll
    for (int t = 0; t < NT; ++t)
      acc[t] = __builtin_amdgcn_wmma_f32_16x16x32_bf16(
          false, a1.v, false, b1[t].v, (short)0, acc[t], false, false);
  }
  // epilogue: final 64 of K
  load_set(k0 + 32, a1, b1);
#pragma unroll
  for (int t = 0; t < NT; ++t)
    acc[t] = __builtin_amdgcn_wmma_f32_16x16x32_bf16(
        false, a0.v, false, b0[t].v, (short)0, acc[t], false, false);
#pragma unroll
  for (int t = 0; t < NT; ++t)
    acc[t] = __builtin_amdgcn_wmma_f32_16x16x32_bf16(
        false, a1.v, false, b1[t].v, (short)0, acc[t], false, false);

#pragma unroll
  for (int t = 0; t < NT; ++t) {
    const int n = n0 + t * 16 + l16;
#pragma unroll
    for (int j = 0; j < 8; ++j) {
      const int m = m0 + j + 8 * half;
      float v = acc[t][j];
      if (EPI == 1) v = softplus_f(v + bias[n]);
      if (EPI == 2) v += resid[(size_t)m * ldr + n];
      C[(size_t)m * ldc + n] = v;
      if (WRITE_BF) Cbf[(size_t)m * ldc + n] = f2bf(v);
    }
  }
}

// ---------------------------------------------------------------------------
// Depthwise causal conv (D_CONV=4) + bias + SiLU; emit f32 and bf16
// xc is the first D_INNER columns of xz (row stride 2*D_INNER)
// ---------------------------------------------------------------------------
__global__ __launch_bounds__(256) void conv_silu_kernel(
    const float* __restrict__ xz, const float* __restrict__ cw,
    const float* __restrict__ cb, float* __restrict__ xs,
    unsigned short* __restrict__ xs_bf) {
  const int idx = blockIdx.x * 256 + threadIdx.x; // over SEQ_L * D_INNER
  const int l   = idx >> 11;
  const int d   = idx & (D_INNER - 1);
  float acc = cb[d];
#pragma unroll
  for (int k = 0; k < D_CONV; ++k) {
    const int lk = l + k - (D_CONV - 1);
    if (lk >= 0) acc += xz[(size_t)lk * (2 * D_INNER) + d] * cw[d * D_CONV + k];
  }
  const float s = silu_f(acc);
  xs[idx]    = s;
  xs_bf[idx] = f2bf(s);
}

// ---------------------------------------------------------------------------
// Selective scan: lane = (channel d, state n). 16 lanes per channel,
// h_{t} = exp(delta*A)*h + delta*B_t*x; y = sum_n h*C_t via shfl-xor tree.
// Also folds in the skip term  + xs * D.
// ---------------------------------------------------------------------------
__global__ __launch_bounds__(256) void scan_kernel(
    const float* __restrict__ delta, // L x D_INNER
    const float* __restrict__ dbc,   // L x XPROJ_PAD (Bm at +64, Cm at +80)
    const float* __restrict__ xs,    // L x D_INNER
    const float* __restrict__ A_log, // D_INNER x 16
    const float* __restrict__ Dvec,  // D_INNER
    float* __restrict__ ybuf)        // L x D_INNER
{
  const int n  = threadIdx.x & 15;
  const int dl = threadIdx.x >> 4;           // 0..15 channels per block
  const int d  = blockIdx.x * 16 + dl;
  const float a  = -__expf(A_log[d * D_STATE + n]);
  const float dd = Dvec[d];
  float h = 0.0f;
  for (int l = 0; l < SEQ_L; ++l) {
    const float dlt = delta[(size_t)l * D_INNER + d];
    const float xv  = xs[(size_t)l * D_INNER + d];
    const float bm  = dbc[(size_t)l * XPROJ_PAD + DT_RANK + n];
    const float cm  = dbc[(size_t)l * XPROJ_PAD + DT_RANK + D_STATE + n];
    h = __expf(dlt * a) * h + dlt * bm * xv;
    float y = h * cm;
    y += __shfl_xor(y, 1, 32);
    y += __shfl_xor(y, 2, 32);
    y += __shfl_xor(y, 4, 32);
    y += __shfl_xor(y, 8, 32);
    if (n == 0) ybuf[(size_t)l * D_INNER + d] = y + xv * dd;
  }
}

// ---------------------------------------------------------------------------
// Gate: yg = y * silu(res); res = xz[:, D_INNER:]; emit bf16 for out_proj
// ---------------------------------------------------------------------------
__global__ __launch_bounds__(256) void gate_kernel(
    const float* __restrict__ ybuf, const float* __restrict__ xz,
    unsigned short* __restrict__ yg) {
  const int idx = blockIdx.x * 256 + threadIdx.x;
  const int l   = idx >> 11;
  const int d   = idx & (D_INNER - 1);
  const float res = xz[(size_t)l * (2 * D_INNER) + D_INNER + d];
  yg[idx] = f2bf(ybuf[idx] * silu_f(res));
}

// ---------------------------------------------------------------------------
extern "C" void kernel_launch(void* const* d_in, const int* in_sizes, int n_in,
                              void* d_out, int out_size, void* d_ws, size_t ws_size,
                              hipStream_t stream) {
  const float* x      = (const float*)d_in[0];
  const float* W_in   = (const float*)d_in[1];
  const float* conv_w = (const float*)d_in[2];
  const float* conv_b = (const float*)d_in[3];
  const float* W_xprj = (const float*)d_in[4];
  const float* W_dt   = (const float*)d_in[5];
  const float* b_dt   = (const float*)d_in[6];
  const float* A_log  = (const float*)d_in[7];
  const float* Dvec   = (const float*)d_in[8];
  const float* W_out  = (const float*)d_in[9];
  const float* norm_w = (const float*)d_in[10];
  float* out = (float*)d_out;

  // carve scratch (256B aligned regions)
  char* ws = (char*)d_ws;
  size_t off = 0;
  auto alloc = [&](size_t bytes) -> void* {
    off = (off + 255) & ~(size_t)255;
    void* p = ws + off;
    off += bytes;
    return p;
  };
  unsigned short* xn_bf   = (unsigned short*)alloc((size_t)SEQ_L * D_MODEL * 2);
  unsigned short* Win_bf  = (unsigned short*)alloc((size_t)2 * D_INNER * D_MODEL * 2);
  unsigned short* Wxp_bf  = (unsigned short*)alloc((size_t)XPROJ_PAD * D_INNER * 2); // zero-padded rows 96..127
  unsigned short* Wdt_bf  = (unsigned short*)alloc((size_t)D_INNER * DT_RANK * 2);
  unsigned short* Wout_bf = (unsigned short*)alloc((size_t)D_MODEL * D_INNER * 2);
  float*          xz      = (float*)alloc((size_t)SEQ_L * 2 * D_INNER * 4);
  float*          xs      = (float*)alloc((size_t)SEQ_L * D_INNER * 4);
  unsigned short* xs_bf   = (unsigned short*)alloc((size_t)SEQ_L * D_INNER * 2);
  float*          dbc     = (float*)alloc((size_t)SEQ_L * XPROJ_PAD * 4);
  unsigned short* dbc_bf  = (unsigned short*)alloc((size_t)SEQ_L * XPROJ_PAD * 2);
  float*          delta   = (float*)alloc((size_t)SEQ_L * D_INNER * 4);
  float*          ybuf    = (float*)alloc((size_t)SEQ_L * D_INNER * 4);
  unsigned short* yg_bf   = (unsigned short*)alloc((size_t)SEQ_L * D_INNER * 2);

  // weight down-converts (bf16); W_xproj zero-padded from 96 to 128 rows
  {
    int n;
    n = 2 * D_INNER * D_MODEL;
    f32_to_bf16_pad_kernel<<<(n + 255) / 256, 256, 0, stream>>>(W_in, Win_bf, n, n);
    n = XPROJ_PAD * D_INNER;
    f32_to_bf16_pad_kernel<<<(n + 255) / 256, 256, 0, stream>>>(W_xprj, Wxp_bf,
                                                                XPROJ_N * D_INNER, n);
    n = D_INNER * DT_RANK;
    f32_to_bf16_pad_kernel<<<(n + 255) / 256, 256, 0, stream>>>(W_dt, Wdt_bf, n, n);
    n = D_MODEL * D_INNER;
    f32_to_bf16_pad_kernel<<<(n + 255) / 256, 256, 0, stream>>>(W_out, Wout_bf, n, n);
  }

  // RMSNorm
  rmsnorm_bf16_kernel<<<SEQ_L, 256, 0, stream>>>(x, norm_w, xn_bf);

  const dim3 gblk(32, 4);
  // in_proj: M=2048, N=4096, K=1024  (grid: 128 x 16, NT=4)
  gemm_wmma_bf16<4, 0, false><<<dim3(SEQ_L / 16, (2 * D_INNER) / 256), gblk, 0, stream>>>(
      xn_bf, D_MODEL, Win_bf, D_MODEL, xz, 2 * D_INNER, nullptr,
      D_MODEL, nullptr, nullptr, 0);

  // depthwise conv + SiLU
  conv_silu_kernel<<<(SEQ_L * D_INNER) / 256, 256, 0, stream>>>(xz, conv_w, conv_b, xs, xs_bf);

  // x_proj: M=2048, N=128 (padded from 96), K=2048 (grid: 128 x 1, NT=2)
  gemm_wmma_bf16<2, 0, true><<<dim3(SEQ_L / 16, 1), gblk, 0, stream>>>(
      xs_bf, D_INNER, Wxp_bf, D_INNER, dbc, XPROJ_PAD, dbc_bf,
      D_INNER, nullptr, nullptr, 0);

  // delta = softplus(dt @ W_dt^T + b_dt): M=2048, N=2048, K=64 (grid: 128 x 8)
  gemm_wmma_bf16<4, 1, false><<<dim3(SEQ_L / 16, D_INNER / 256), gblk, 0, stream>>>(
      dbc_bf, XPROJ_PAD, Wdt_bf, DT_RANK, delta, D_INNER, nullptr,
      DT_RANK, b_dt, nullptr, 0);

  // selective scan (+ xs * D)
  scan_kernel<<<D_INNER / 16, 256, 0, stream>>>(delta, dbc, xs, A_log, Dvec, ybuf);

  // gate with silu(res)
  gate_kernel<<<(SEQ_L * D_INNER) / 256, 256, 0, stream>>>(ybuf, xz, yg_bf);

  // out_proj + residual: M=2048, N=1024, K=2048 (grid: 128 x 4)
  gemm_wmma_bf16<4, 2, false><<<dim3(SEQ_L / 16, D_MODEL / 256), gblk, 0, stream>>>(
      yg_bf, D_INNER, Wout_bf, D_INNER, out, D_MODEL, nullptr,
      D_INNER, nullptr, x, D_MODEL);

  (void)in_sizes; (void)n_in; (void)out_size; (void)ws_size;
}
